// DecoderLayer_4750233829964
// MI455X (gfx1250) — matrix operand order repaired and checked by reading, not compiled, not run on hardware
//
#include <hip/hip_runtime.h>

// ---------------------------------------------------------------------------
// Transformer decoder layer for gfx1250 (MI455X).
// fp32 I/O, f16 WMMA (v_wmma_f32_16x16x32_f16) with f32 accumulation.
// 128x128 (or 128x64) block tiles, double-buffered LDS + register prefetch:
// one barrier per K-step so staging VALU/DS co-executes with the WMMA pipe.
// ---------------------------------------------------------------------------

typedef __attribute__((ext_vector_type(16))) _Float16 v16h;
typedef __attribute__((ext_vector_type(2)))  __fp16   fp16x2;
typedef __attribute__((ext_vector_type(8)))  float    v8f;

#define BDIM 256
#define BK 32
#define LDSW 17   // dwords per LDS row: 32 halves (16 dwords) + 1 pad dword

// Problem constants (match reference)
#define TB 4
#define TS 1024
#define TE 1024
#define TD 1024
#define THD 16
#define TDH 64
#define TDFF 4096

struct GemmP {
  const float* A; const float* B; float* C; const float* bias;
  int M, N, K;
  int lda, ldb, ldc;
  int relu, nheads;
  float scale;
  const float* mask; long maskRow, maskBatch;
  long bA, hA, bB, hB, bC, hC;   // z = b*nheads + h pointer strides (elements)
};

union PKU { fp16x2 h; unsigned int u; };
__device__ __forceinline__ unsigned int pkrtz(float a, float b) {
  PKU p; p.h = __builtin_amdgcn_cvt_pkrtz(a, b);   // v_cvt_pk_rtz_f16_f32
  return p.u;
}

template <int BN, bool TRANSB>
__global__ __launch_bounds__(BDIM) void gemm_wmma(GemmP p) {
  constexpr int WNT   = BN / 32;           // 16-wide n-tiles per wave (2 waveN groups)
  constexpr int LOGBN = (BN == 128) ? 7 : 6;
  constexpr int NB    = BN / 16;           // k-pair loads/thread (non-trans B)
  constexpr int NBT   = BN / 32;           // float4 loads/thread (trans B)
  constexpr int SA_SZ = 128 * LDSW;
  constexpr int SB_SZ = BN * LDSW;

  __shared__ unsigned int sA[2 * SA_SZ];   // [buf][m][k/2] packed f16
  __shared__ unsigned int sB[2 * SB_SZ];   // [buf][n][k/2] packed f16 (B transposed)

  const int tid  = threadIdx.x;
  const int lane = tid & 31, wid = tid >> 5;
  const int lh   = lane >> 4, l16 = lane & 15;
  const int waveM = wid & 3, waveN = wid >> 2;

  const int z  = blockIdx.z;
  const int bz = z / p.nheads, hz = z % p.nheads;
  const float* A  = p.A + (long)bz * p.bA + (long)hz * p.hA;
  const float* Bm = p.B + (long)bz * p.bB + (long)hz * p.hB;
  float*       C  = p.C + (long)bz * p.bC + (long)hz * p.hC;

  const int m0 = blockIdx.y * 128;
  const int n0 = blockIdx.x * BN;

  v8f acc[2][WNT] = {};

  float4 ga[4];        // A panel prefetch (128x32 fp32 via 1024 float4)
  float4 gbt[NBT];     // B panel prefetch, transB layout
  float2 gbn[NB];      // B panel prefetch, normal layout (k-pairs)

  auto load_panel = [&](int kt) {
#pragma unroll
    for (int t = 0; t < 4; ++t) {
      int lin = tid + t * BDIM;
      int row = lin >> 3, f4 = lin & 7;
      ga[t] = *reinterpret_cast<const float4*>(
          A + (size_t)(m0 + row) * p.lda + kt + f4 * 4);
    }
    if constexpr (TRANSB) {
#pragma unroll
      for (int t = 0; t < NBT; ++t) {
        int lin = tid + t * BDIM;
        int n = lin >> 3, f4 = lin & 7;
        gbt[t] = *reinterpret_cast<const float4*>(
            Bm + (size_t)(n0 + n) * p.ldb + kt + f4 * 4);
      }
    } else {
#pragma unroll
      for (int t = 0; t < NB; ++t) {
        int lin = tid + t * BDIM;
        int n = lin & (BN - 1), kp = lin >> LOGBN;
        gbn[t].x = Bm[(size_t)(kt + 2 * kp)     * p.ldb + n0 + n];
        gbn[t].y = Bm[(size_t)(kt + 2 * kp + 1) * p.ldb + n0 + n];
      }
    }
  };

  auto stage_lds = [&](int buf) {
    unsigned int* dA = sA + buf * SA_SZ;
    unsigned int* dB = sB + buf * SB_SZ;
#pragma unroll
    for (int t = 0; t < 4; ++t) {
      int lin = tid + t * BDIM;
      int row = lin >> 3, f4 = lin & 7;
      dA[row * LDSW + f4 * 2]     = pkrtz(ga[t].x, ga[t].y);
      dA[row * LDSW + f4 * 2 + 1] = pkrtz(ga[t].z, ga[t].w);
    }
    if constexpr (TRANSB) {
#pragma unroll
      for (int t = 0; t < NBT; ++t) {
        int lin = tid + t * BDIM;
        int n = lin >> 3, f4 = lin & 7;
        dB[n * LDSW + f4 * 2]     = pkrtz(gbt[t].x, gbt[t].y);
        dB[n * LDSW + f4 * 2 + 1] = pkrtz(gbt[t].z, gbt[t].w);
      }
    } else {
#pragma unroll
      for (int t = 0; t < NB; ++t) {
        int lin = tid + t * BDIM;
        int n = lin & (BN - 1), kp = lin >> LOGBN;
        dB[n * LDSW + kp] = pkrtz(gbn[t].x, gbn[t].y);
      }
    }
  };

  // --- pipeline prologue: LDS buf0 <- panel0, regs <- panel1
  load_panel(0);
  stage_lds(0);
  if (BK < p.K) load_panel(BK);
  __syncthreads();

  for (int kt = 0; kt < p.K; kt += BK) {
    const int cur = (kt >> 5) & 1;

    // 1) fragment DS loads from current buffer (issued before the DS stores
    //    below, so the pre-WMMA wait is a partial dscnt, not 0)
    union F { v16h v; unsigned int u[8]; };
    F afr[2], bfr[WNT];
    const unsigned int* bA = sA + cur * SA_SZ;
    const unsigned int* bB = sB + cur * SB_SZ;
#pragma unroll
    for (int tm = 0; tm < 2; ++tm) {
      const unsigned int* pa = bA + (waveM * 32 + tm * 16 + l16) * LDSW;
#pragma unroll
      for (int j = 0; j < 4; ++j) {   // k = lh*16 + e
        afr[tm].u[j]     = pa[lh * 8 + j];
        afr[tm].u[4 + j] = pa[lh * 8 + 4 + j];
      }
    }
#pragma unroll
    for (int tn = 0; tn < WNT; ++tn) {
      const unsigned int* pb = bB + (waveN * 16 * WNT + tn * 16 + l16) * LDSW;
#pragma unroll
      for (int j = 0; j < 8; ++j)     // k = lh*16 + e
        bfr[tn].u[j] = pb[lh * 8 + j];
    }

    // 2) stage next panel (regs) into the other buffer — overlaps WMMA below
    if (kt + BK < p.K) stage_lds(cur ^ 1);
    // 3) prefetch panel kt+2*BK into regs (consumed by next iteration's stage)
    if (kt + 2 * BK < p.K) load_panel(kt + 2 * BK);

    // 4) matrix pipe
#pragma unroll
    for (int tm = 0; tm < 2; ++tm)
#pragma unroll
      for (int tn = 0; tn < WNT; ++tn)
        acc[tm][tn] = __builtin_amdgcn_wmma_f32_16x16x32_f16(
            false, afr[tm].v, false, bfr[tn].v,
            (short)0, acc[tm][tn], false, false);

    // 5) single barrier: protects buf[cur^1] stores vs next-iter reads, and
    //    buf[cur] reads vs its restaging in the next iteration
    __syncthreads();
  }

  // --- epilogue: scale, bias, additive mask (* -1e9), relu, fp32 store
#pragma unroll
  for (int tm = 0; tm < 2; ++tm) {
#pragma unroll
    for (int tn = 0; tn < WNT; ++tn) {
      const int n = n0 + waveN * 16 * WNT + tn * 16 + l16;
#pragma unroll
      for (int r = 0; r < 8; ++r) {
        const int m = m0 + waveM * 32 + tm * 16 + lh * 8 + r;
        float v = acc[tm][tn][r] * p.scale;
        if (p.bias) v += p.bias[n];
        if (p.mask) v += p.mask[(long)bz * p.maskBatch + (long)m * p.maskRow + n] * (-1e9f);
        if (p.relu) v = fmaxf(v, 0.0f);
        C[(size_t)m * p.ldc + n] = v;
      }
    }
  }
}

// One wave per row of 1024; 8 rows per block. In-place numerically-stable softmax.
__global__ __launch_bounds__(256) void softmax_rows1024(float* data) {
  const int lane = threadIdx.x & 31, wid = threadIdx.x >> 5;
  float* p = data + ((long)blockIdx.x * 8 + wid) * 1024;
  float v[32];
  float mx = -3.4e38f;
#pragma unroll
  for (int j = 0; j < 32; ++j) { v[j] = p[lane + (j << 5)]; mx = fmaxf(mx, v[j]); }
#pragma unroll
  for (int off = 16; off > 0; off >>= 1) mx = fmaxf(mx, __shfl_xor(mx, off, 32));
  float s = 0.f;
#pragma unroll
  for (int j = 0; j < 32; ++j) { v[j] = __expf(v[j] - mx); s += v[j]; }
#pragma unroll
  for (int off = 16; off > 0; off >>= 1) s += __shfl_xor(s, off, 32);
  const float inv = 1.f / s;
#pragma unroll
  for (int j = 0; j < 32; ++j) p[lane + (j << 5)] = v[j] * inv;
}

// out = LayerNorm(x + r) * g + b, row length 1024, one block (256 thr) per row.
__global__ __launch_bounds__(256) void add_ln1024(const float* x, const float* r,
                                                  const float* g, const float* b,
                                                  float* out) {
  __shared__ float red[512];
  const int tid = threadIdx.x;
  const size_t base = (size_t)blockIdx.x * 1024;
  float v[4];
  float s = 0.f, ss = 0.f;
#pragma unroll
  for (int j = 0; j < 4; ++j) {
    const int c = tid + j * 256;
    const float t = x[base + c] + r[base + c];
    v[j] = t; s += t; ss += t * t;
  }
  red[tid] = s; red[256 + tid] = ss;
  __syncthreads();
  for (int st = 128; st > 0; st >>= 1) {
    if (tid < st) { red[tid] += red[tid + st]; red[256 + tid] += red[256 + tid + st]; }
    __syncthreads();
  }
  const float mu   = red[0] * (1.f / 1024.f);
  const float var  = red[256] * (1.f / 1024.f) - mu * mu;
  const float rstd = rsqrtf(var + 1e-6f);
#pragma unroll
  for (int j = 0; j < 4; ++j) {
    const int c = tid + j * 256;
    out[base + c] = (v[j] - mu) * rstd * g[c] + b[c];
  }
}

static inline void launch_gemm(hipStream_t st, const float* A, const float* B, float* C,
                               const float* bias, int M, int N, int K,
                               int lda, int ldb, int ldc, int bn, int transB, int relu,
                               float scale,
                               const float* mask, long maskRow, long maskBatch,
                               int batch, int nheads,
                               long bA, long hA, long bB, long hB, long bC, long hC) {
  GemmP p;
  p.A = A; p.B = B; p.C = C; p.bias = bias;
  p.M = M; p.N = N; p.K = K; p.lda = lda; p.ldb = ldb; p.ldc = ldc;
  p.relu = relu; p.nheads = nheads; p.scale = scale;
  p.mask = mask; p.maskRow = maskRow; p.maskBatch = maskBatch;
  p.bA = bA; p.hA = hA; p.bB = bB; p.hB = hB; p.bC = bC; p.hC = hC;
  dim3 grid((unsigned)(N / bn), (unsigned)(M / 128), (unsigned)batch);
  if (bn == 128) {
    if (transB)
      hipLaunchKernelGGL((gemm_wmma<128, true>),  grid, dim3(BDIM), 0, st, p);
    else
      hipLaunchKernelGGL((gemm_wmma<128, false>), grid, dim3(BDIM), 0, st, p);
  } else {
    hipLaunchKernelGGL((gemm_wmma<64, false>), grid, dim3(BDIM), 0, st, p);
  }
}

extern "C" void kernel_launch(void* const* d_in, const int* in_sizes, int n_in,
                              void* d_out, int out_size, void* d_ws, size_t ws_size,
                              hipStream_t stream) {
  (void)in_sizes; (void)n_in; (void)out_size; (void)ws_size;
  const float* dec  = (const float*)d_in[0];
  const float* enc  = (const float*)d_in[1];
  const float* look = (const float*)d_in[2];   // [1,1,S,S]
  const float* pad  = (const float*)d_in[3];   // [B,1,1,E]
  const float* w1q = (const float*)d_in[4];  const float* b1q = (const float*)d_in[5];
  const float* w1k = (const float*)d_in[6];  const float* b1k = (const float*)d_in[7];
  const float* w1v = (const float*)d_in[8];  const float* b1v = (const float*)d_in[9];
  const float* w1o = (const float*)d_in[10]; const float* b1o = (const float*)d_in[11];
  const float* w2q = (const float*)d_in[12]; const float* b2q = (const float*)d_in[13];
  const float* w2k = (const float*)d_in[14]; const float* b2k = (const float*)d_in[15];
  const float* w2v = (const float*)d_in[16]; const float* b2v = (const float*)d_in[17];
  const float* w2o = (const float*)d_in[18]; const float* b2o = (const float*)d_in[19];
  const float* fw1 = (const float*)d_in[20]; const float* fb1 = (const float*)d_in[21];
  const float* fw2 = (const float*)d_in[22]; const float* fb2 = (const float*)d_in[23];
  const float* g1 = (const float*)d_in[24]; const float* be1 = (const float*)d_in[25];
  const float* g2 = (const float*)d_in[26]; const float* be2 = (const float*)d_in[27];
  const float* g3 = (const float*)d_in[28]; const float* be3 = (const float*)d_in[29];

  const long M  = (long)TB * TS;   // 4096 rows of activations
  const long MD = M * TD;          // 4Mi elements

  float* out   = (float*)d_out;
  float* attn1 = out + MD;                               // [B,H,S,S]
  float* attn2 = attn1 + (long)TB * THD * TS * TS;       // [B,H,S,E]

  float* ws   = (float*)d_ws;
  float* q    = ws;            // [M,D]
  float* kbuf = q + MD;
  float* vbuf = kbuf + MD;
  float* ctx  = vbuf + MD;
  float* tmp  = ctx + MD;
  float* qry  = tmp + MD;
  float* ybuf = qry + MD;
  float* hid  = ybuf + MD;     // [M,DFF]

  const float scl = 0.125f;    // 1/sqrt(64)

  // ---- MHA1: projections ----
  launch_gemm(stream, dec, w1q, q,    b1q, M, TD, TD, TD, TD, TD, 128, 0, 0, 1.f,
              nullptr, 0, 0, 1, 1, 0, 0, 0, 0, 0, 0);
  launch_gemm(stream, dec, w1k, kbuf, b1k, M, TD, TD, TD, TD, TD, 128, 0, 0, 1.f,
              nullptr, 0, 0, 1, 1, 0, 0, 0, 0, 0, 0);
  launch_gemm(stream, dec, w1v, vbuf, b1v, M, TD, TD, TD, TD, TD, 128, 0, 0, 1.f,
              nullptr, 0, 0, 1, 1, 0, 0, 0, 0, 0, 0);
  // scores1 = Q K^T * 1/8 + look*(-1e9), batched over B*H heads
  launch_gemm(stream, q, kbuf, attn1, nullptr, TS, TS, TDH, TD, TD, TS, 128, 1, 0, scl,
              look, TS, 0, TB * THD, THD,
              (long)TS * TD, TDH, (long)TS * TD, TDH,
              (long)THD * TS * TS, (long)TS * TS);
  softmax_rows1024<<<(TB * THD * TS) / 8, 256, 0, stream>>>(attn1);
  // ctx1 = attn1 @ V  (writes head slices back into [M,D] layout)
  launch_gemm(stream, attn1, vbuf, ctx, nullptr, TS, TDH, TS, TS, TD, TD, 64, 0, 0, 1.f,
              nullptr, 0, 0, TB * THD, THD,
              (long)THD * TS * TS, (long)TS * TS,
              (long)TS * TD, TDH, (long)TS * TD, TDH);
  launch_gemm(stream, ctx, w1o, tmp, b1o, M, TD, TD, TD, TD, TD, 128, 0, 0, 1.f,
              nullptr, 0, 0, 1, 1, 0, 0, 0, 0, 0, 0);
  add_ln1024<<<M, 256, 0, stream>>>(dec, tmp, g1, be1, qry);

  // ---- MHA2 (cross-attention) ----
  launch_gemm(stream, qry, w2q, q,    b2q, M, TD, TD, TD, TD, TD, 128, 0, 0, 1.f,
              nullptr, 0, 0, 1, 1, 0, 0, 0, 0, 0, 0);
  launch_gemm(stream, enc, w2k, kbuf, b2k, M, TD, TD, TD, TD, TD, 128, 0, 0, 1.f,
              nullptr, 0, 0, 1, 1, 0, 0, 0, 0, 0, 0);
  launch_gemm(stream, enc, w2v, vbuf, b2v, M, TD, TD, TD, TD, TD, 128, 0, 0, 1.f,
              nullptr, 0, 0, 1, 1, 0, 0, 0, 0, 0, 0);
  launch_gemm(stream, q, kbuf, attn2, nullptr, TS, TE, TDH, TD, TD, TE, 128, 1, 0, scl,
              pad, 0, TE, TB * THD, THD,
              (long)TS * TD, TDH, (long)TE * TD, TDH,
              (long)THD * TS * TE, (long)TS * TE);
  softmax_rows1024<<<(TB * THD * TS) / 8, 256, 0, stream>>>(attn2);
  launch_gemm(stream, attn2, vbuf, ctx, nullptr, TS, TDH, TE, TE, TD, TD, 64, 0, 0, 1.f,
              nullptr, 0, 0, TB * THD, THD,
              (long)THD * TS * TE, (long)TS * TE,
              (long)TE * TD, TDH, (long)TE * TD, TDH);
  launch_gemm(stream, ctx, w2o, tmp, b2o, M, TD, TD, TD, TD, TD, 128, 0, 0, 1.f,
              nullptr, 0, 0, 1, 1, 0, 0, 0, 0, 0, 0);
  add_ln1024<<<M, 256, 0, stream>>>(tmp, qry, g2, be2, ybuf);

  // ---- FFN ----
  launch_gemm(stream, ybuf, fw1, hid, fb1, M, TDFF, TD, TD, TDFF, TDFF, 128, 0, 1, 1.f,
              nullptr, 0, 0, 1, 1, 0, 0, 0, 0, 0, 0);
  launch_gemm(stream, hid, fw2, tmp, fb2, M, TD, TDFF, TDFF, TD, TD, 128, 0, 0, 1.f,
              nullptr, 0, 0, 1, 1, 0, 0, 0, 0, 0, 0);
  add_ln1024<<<M, 256, 0, stream>>>(ybuf, tmp, g3, be3, out);
}
